// multi_scale_attention_encoder_2241972929092
// MI455X (gfx1250) — matrix operand order repaired and checked by reading, not compiled, not run on hardware
//
#include <hip/hip_runtime.h>
#include <math.h>

// ---------------------------------------------------------------------------
// MI455X / gfx1250, wave32. All dense matmuls (q/out/FF projections, scores,
// attn*V) run on v_wmma_f32_16x16x32_f16 with f32 accumulation. K/V are kept
// in f16 in the workspace; K/V tiles are staged to LDS once per key tile
// (heads-inner flash attention) with global_prefetch_b8 on the next tile.
// ---------------------------------------------------------------------------

typedef __attribute__((ext_vector_type(16))) _Float16 v16h;
typedef __attribute__((ext_vector_type(8)))  float    v8f;

#define WMMA_F16(a, b, c) \
  __builtin_amdgcn_wmma_f32_16x16x32_f16(false, (a), false, (b), (short)0, (c), false, false)

// ---------------- WMMA fragment builders (contiguous k = 16*(lane/16) + s in both A and B) ----

// A fragment (16x32 f16) from row-major f16 buffer, rows = M, K range [kofs,kofs+32)
static __device__ inline v16h frag_a(const _Float16* src, int ld, int kofs, int lane) {
  int row = lane & 15;
  int kb  = kofs + ((lane & 16) ? 16 : 0);
  const _Float16* p = src + row * ld + kb;
  v16h a;
#pragma unroll
  for (int i = 0; i < 8; ++i) a[i] = p[i];
#pragma unroll
  for (int i = 0; i < 8; ++i) a[8 + i] = p[8 + i];
  return a;
}

// A fragment where only K=0..15 are valid (columns dofs..dofs+15), K=16..31 zero
static __device__ inline v16h frag_a_pad16(const _Float16* src, int ld, int dofs, int lane) {
  int row = lane & 15;
  bool lo = lane < 16;
  const _Float16* p = src + row * ld + dofs;
  v16h a;
#pragma unroll
  for (int i = 0; i < 8; ++i) a[i] = lo ? p[i] : (_Float16)0.0f;
#pragma unroll
  for (int i = 0; i < 8; ++i) a[8 + i] = lo ? p[8 + i] : (_Float16)0.0f;
  return a;
}

// B fragment (32x16) from "weight" layout: element(k,n) = src[(nofs+n)*ld + kofs + k]
static __device__ inline v16h frag_b_wt(const _Float16* src, int ld, int nofs, int kofs, int lane) {
  int col = lane & 15;
  int kb  = kofs + ((lane & 16) ? 16 : 0);
  const _Float16* p = src + (size_t)(nofs + col) * ld + kb;
  v16h b;
#pragma unroll
  for (int i = 0; i < 16; ++i) b[i] = p[i];
  return b;
}

// B fragment, K padded to 16: element(k,n) = (k<16) ? src[n*ld + dofs + k] : 0
static __device__ inline v16h frag_b_nk_pad16(const _Float16* src, int ld, int dofs, int lane) {
  int col = lane & 15;
  bool lo = lane < 16;
  const _Float16* p = src + col * ld + dofs;
  v16h b;
#pragma unroll
  for (int i = 0; i < 16; ++i) b[i] = lo ? p[i] : (_Float16)0.0f;
  return b;
}

// B fragment, K-major source, K padded to 16: element(k,n) = (k<16) ? src[k*ld + dofs + n] : 0
static __device__ inline v16h frag_b_km_pad16(const _Float16* src, int ld, int dofs, int lane) {
  int col = lane & 15;
  bool lo = lane < 16;
  v16h b;
#pragma unroll
  for (int i = 0; i < 16; ++i) b[i] = lo ? src[i * ld + dofs + col] : (_Float16)0.0f;
  return b;
}

static __device__ inline float gelu_exact(float x) {
  return 0.5f * x * (1.0f + erff(x * 0.70710678118654752f));
}

// ---------------------------------------------------------------------------
// FPS: one workgroup per batch; sequential argmax scan with block reduction.
// ---------------------------------------------------------------------------
static __device__ void fps_run(float* px, float* py, float* pz, float* dist,
                               float* selx, float* sely, float* selz,
                               float* rv, int* ri, int* s_last,
                               int N, int K, int tid) {
  if (tid == 0) { selx[0] = px[0]; sely[0] = py[0]; selz[0] = pz[0]; *s_last = 0; }
  __syncthreads();
  for (int step = 1; step < K; ++step) {
    int last = *s_last;
    float lx = px[last], ly = py[last], lz = pz[last];
    float bestv = -1.0f;
    int   besti = 0x7fffffff;
    for (int i = tid; i < N; i += 256) {
      float dx = px[i] - lx, dy = py[i] - ly, dz = pz[i] - lz;
      float d  = dx * dx + dy * dy + dz * dz;
      float nd = fminf(dist[i], d);
      dist[i]  = nd;
      if (nd > bestv) { bestv = nd; besti = i; }   // strict > keeps first index
    }
    rv[tid] = bestv; ri[tid] = besti;
    __syncthreads();
    for (int s = 128; s > 0; s >>= 1) {
      if (tid < s) {
        float ov = rv[tid + s]; int oi = ri[tid + s];
        if (ov > rv[tid] || (ov == rv[tid] && oi < ri[tid])) { rv[tid] = ov; ri[tid] = oi; }
      }
      __syncthreads();
    }
    if (tid == 0) {
      int nxt = ri[0];
      *s_last = nxt;
      selx[step] = px[nxt]; sely[step] = py[nxt]; selz[step] = pz[nxt];
    }
    __syncthreads();
  }
}

__global__ void __launch_bounds__(256) fps_kernel(const float* __restrict__ xyz,
                                                  float* __restrict__ lvl1,
                                                  float* __restrict__ lvl2) {
  __shared__ float px[2048], py[2048], pz[2048], dist[2048];
  __shared__ float selx[512], sely[512], selz[512];
  __shared__ float rv[256];
  __shared__ int   ri[256];
  __shared__ int   s_last;
  const int b = blockIdx.x, tid = threadIdx.x;

  // level 1: 2048 -> 512
  const float* src = xyz + (size_t)b * 2048 * 3;
  for (int i = tid; i < 2048; i += 256) {
    px[i] = src[i * 3]; py[i] = src[i * 3 + 1]; pz[i] = src[i * 3 + 2];
    dist[i] = 1e10f;
  }
  __syncthreads();
  fps_run(px, py, pz, dist, selx, sely, selz, rv, ri, &s_last, 2048, 512, tid);
  for (int i = tid; i < 512; i += 256) {
    float* d = lvl1 + ((size_t)b * 512 + i) * 3;
    d[0] = selx[i]; d[1] = sely[i]; d[2] = selz[i];
  }
  __syncthreads();

  // level 2: 512 -> 256 (from level-1 points)
  for (int i = tid; i < 512; i += 256) {
    px[i] = selx[i]; py[i] = sely[i]; pz[i] = selz[i]; dist[i] = 1e10f;
  }
  __syncthreads();
  fps_run(px, py, pz, dist, selx, sely, selz, rv, ri, &s_last, 512, 256, tid);
  for (int i = tid; i < 256; i += 256) {
    float* d = lvl2 + ((size_t)b * 256 + i) * 3;
    d[0] = selx[i]; d[1] = sely[i]; d[2] = selz[i];
  }
}

// ---------------------------------------------------------------------------
// mlp_conv: 3 -> 32 (relu) -> 64 per point. K=3 is below WMMA shapes: scalar.
// ---------------------------------------------------------------------------
__global__ void __launch_bounds__(256) mlp_kernel(const float* __restrict__ x,
                                                  const float* __restrict__ W1,
                                                  const float* __restrict__ b1,
                                                  const float* __restrict__ W2,
                                                  const float* __restrict__ b2,
                                                  float* __restrict__ y, int total) {
  __shared__ float sW1[96], sb1[32], sW2[2048], sb2[64];
  const int tid = threadIdx.x;
  for (int i = tid; i < 96; i += 256) sW1[i] = W1[i];
  for (int i = tid; i < 2048; i += 256) sW2[i] = W2[i];
  if (tid < 32) sb1[tid] = b1[tid];
  if (tid < 64) sb2[tid] = b2[tid];
  __syncthreads();
  const int idx = blockIdx.x * 256 + tid;
  if (idx >= total) return;
  const float x0 = x[(size_t)idx * 3], x1 = x[(size_t)idx * 3 + 1], x2 = x[(size_t)idx * 3 + 2];
  float h[32];
#pragma unroll
  for (int o = 0; o < 32; ++o)
    h[o] = fmaxf(0.0f, sW1[o * 3] * x0 + sW1[o * 3 + 1] * x1 + sW1[o * 3 + 2] * x2 + sb1[o]);
  float* yp = y + (size_t)idx * 64;
  for (int o = 0; o < 64; ++o) {
    float acc = sb2[o];
    const float* w = sW2 + o * 32;
#pragma unroll
    for (int c = 0; c < 32; ++c) acc += w[c] * h[c];
    yp[o] = acc;
  }
}

// ---------------------------------------------------------------------------
// y = LayerNorm(x @ projW^T + projb) with n13 params. One point per thread.
// ---------------------------------------------------------------------------
__global__ void __launch_bounds__(256) proj_ln_kernel(const float* __restrict__ x,
                                                      const float* __restrict__ W,
                                                      const float* __restrict__ bias,
                                                      const float* __restrict__ g,
                                                      const float* __restrict__ bt,
                                                      float* __restrict__ y, int total) {
  __shared__ float sW[64 * 64];
  __shared__ float sb[64], sg[64], sbt[64];
  const int tid = threadIdx.x;
  for (int i = tid; i < 4096; i += 256) sW[i] = W[i];
  if (tid < 64) { sb[tid] = bias[tid]; sg[tid] = g[tid]; sbt[tid] = bt[tid]; }
  __syncthreads();
  const int idx = blockIdx.x * 256 + tid;
  if (idx >= total) return;
  float xin[64];
  const float* xp = x + (size_t)idx * 64;
#pragma unroll
  for (int c = 0; c < 64; ++c) xin[c] = xp[c];
  float o[64];
  float mean = 0.0f;
  for (int oi = 0; oi < 64; ++oi) {
    float acc = sb[oi];
    const float* w = sW + oi * 64;
#pragma unroll
    for (int c = 0; c < 64; ++c) acc += w[c] * xin[c];
    o[oi] = acc; mean += acc;
  }
  mean *= (1.0f / 64.0f);
  float var = 0.0f;
#pragma unroll
  for (int oi = 0; oi < 64; ++oi) { float d = o[oi] - mean; var += d * d; }
  var *= (1.0f / 64.0f);
  const float rstd = rsqrtf(var + 1e-5f);
  float* yp = y + (size_t)idx * 64;
#pragma unroll
  for (int oi = 0; oi < 64; ++oi) yp[oi] = (o[oi] - mean) * rstd * sg[oi] + sbt[oi];
}

// ---------------------------------------------------------------------------
// K/V projection of the key-side tensor; emits f16 so the attention inner
// loop streams half the bytes and needs no conversion VALU.
// ---------------------------------------------------------------------------
__global__ void __launch_bounds__(256) kv_kernel(const float* __restrict__ y,
                                                 const float* __restrict__ inW,
                                                 const float* __restrict__ inb,
                                                 _Float16* __restrict__ K,
                                                 _Float16* __restrict__ V, int total) {
  __shared__ float sW[64 * 64];
  __shared__ float sb[64];
  const int tid = threadIdx.x;
  const int idx = blockIdx.x * 256 + tid;
  float xin[64];
  if (idx < total) {
    const float* xp = y + (size_t)idx * 64;
#pragma unroll
    for (int c = 0; c < 64; ++c) xin[c] = xp[c];
  }
  // ---- K (rows 64..127 of in_W)
  for (int i = tid; i < 4096; i += 256) sW[i] = inW[4096 + i];
  if (tid < 64) sb[tid] = inb[64 + tid];
  __syncthreads();
  if (idx < total) {
    _Float16* kp = K + (size_t)idx * 64;
    for (int oi = 0; oi < 64; ++oi) {
      float acc = sb[oi];
      const float* w = sW + oi * 64;
#pragma unroll
      for (int c = 0; c < 64; ++c) acc += w[c] * xin[c];
      kp[oi] = (_Float16)acc;
    }
  }
  __syncthreads();
  // ---- V (rows 128..191 of in_W)
  for (int i = tid; i < 4096; i += 256) sW[i] = inW[8192 + i];
  if (tid < 64) sb[tid] = inb[128 + tid];
  __syncthreads();
  if (idx < total) {
    _Float16* vp = V + (size_t)idx * 64;
    for (int oi = 0; oi < 64; ++oi) {
      float acc = sb[oi];
      const float* w = sW + oi * 64;
#pragma unroll
      for (int c = 0; c < 64; ++c) acc += w[c] * xin[c];
      vp[oi] = (_Float16)acc;
    }
  }
}

// ---------------------------------------------------------------------------
// Fused cross-attention + FF block. One wave (32 threads) per 16-query tile.
// Heads-inner flash attention: each f16 K/V tile staged to LDS once (b128
// copies), next tile prefetched (global_prefetch_b8) while WMMAs run.
// ---------------------------------------------------------------------------
__global__ void __launch_bounds__(32) attn_kernel(
    const float* __restrict__ yq,
    const _Float16* __restrict__ kg, const _Float16* __restrict__ vg,
    const _Float16* __restrict__ wqh, const float* __restrict__ bq,
    const _Float16* __restrict__ owh, const float* __restrict__ ob,
    const float* __restrict__ g12, const float* __restrict__ b12,
    const _Float16* __restrict__ l1h, const float* __restrict__ l1b,
    const _Float16* __restrict__ l2h, const float* __restrict__ l2b,
    float* __restrict__ out, int Lq, int Lk) {
  __shared__ _Float16 sX[16 * 64];     // f16 staging: x1 tile -> attn out -> x1'
  __shared__ float    sXf[16 * 64];    // f32 residual copy (x1, then x1')
  __shared__ _Float16 sQ[16 * 64];     // scaled q
  __shared__ _Float16 sK[16 * 64];     // key tile (f16)
  __shared__ _Float16 sV[16 * 64];     // value tile (f16)
  __shared__ _Float16 sP[4 * 16 * 16]; // per-head softmax probability tiles
  __shared__ float    sO[16 * 64];     // out-proj + residual (pre-LN)
  __shared__ _Float16 sH[16 * 1024];   // gelu hidden

  const int lane   = threadIdx.x;
  const int col    = lane & 15;
  const int rbase  = (lane & 16) ? 8 : 0;
  const int qtiles = Lq >> 4;
  const int bb     = blockIdx.x / qtiles;
  const int q0     = (blockIdx.x % qtiles) << 4;

  // ---- load x1 tile (already proj+LN'd)
  {
    const float* src = yq + ((size_t)bb * Lq + q0) * 64;
    for (int i = lane; i < 16 * 64; i += 32) {
      float v = src[i];
      sXf[i] = v;
      sX[i]  = (_Float16)v;
    }
  }
  __syncthreads();

  // ---- q = (x1 @ Wq^T + bq) * 1/sqrt(dh)
  {
    v16h a0 = frag_a(sX, 64, 0, lane);
    v16h a1 = frag_a(sX, 64, 32, lane);
#pragma unroll
    for (int t = 0; t < 4; ++t) {
      v8f c = {};
      c = WMMA_F16(a0, frag_b_wt(wqh, 64, t * 16, 0, lane), c);
      c = WMMA_F16(a1, frag_b_wt(wqh, 64, t * 16, 32, lane), c);
      float bqv = bq[t * 16 + col];
#pragma unroll
      for (int r = 0; r < 8; ++r)
        sQ[(rbase + r) * 64 + t * 16 + col] = (_Float16)((c[r] + bqv) * 0.25f);
    }
  }
  __syncthreads();

  // ---- flash attention, heads inner (dh=16 padded to K=32); result -> sX
  const int ktiles = Lk >> 4;
  v16h aq[4];
  v8f  acc[4];
  float m[4][8], l[4][8];
#pragma unroll
  for (int h = 0; h < 4; ++h) {
    aq[h]  = frag_a_pad16(sQ, 64, h * 16, lane);
    acc[h] = (v8f){};
#pragma unroll
    for (int r = 0; r < 8; ++r) { m[h][r] = -1e30f; l[h][r] = 0.0f; }
  }

  for (int kt = 0; kt < ktiles; ++kt) {
    __syncthreads();  // protect sK/sV/sP from previous iteration's readers
    {
      const uint4* ks = (const uint4*)(kg + ((size_t)bb * Lk + kt * 16) * 64);
      const uint4* vs = (const uint4*)(vg + ((size_t)bb * Lk + kt * 16) * 64);
      uint4* dk = (uint4*)sK;
      uint4* dv = (uint4*)sV;
#pragma unroll
      for (int i = lane; i < 128; i += 32) { dk[i] = ks[i]; dv[i] = vs[i]; }
      if (kt + 1 < ktiles) {
        // prefetch next K/V tile (2KB each) while this tile is consumed
        const char* pk = (const char*)(kg + ((size_t)bb * Lk + (kt + 1) * 16) * 64);
        const char* pv = (const char*)(vg + ((size_t)bb * Lk + (kt + 1) * 16) * 64);
        __builtin_prefetch(pk + lane * 64, 0, 0);
        __builtin_prefetch(pv + lane * 64, 0, 0);
      }
    }
    __syncthreads();
#pragma unroll
    for (int h = 0; h < 4; ++h) {
      v8f zero = {};
      v8f s = WMMA_F16(aq[h], frag_b_nk_pad16(sK, 64, h * 16, lane), zero);
      _Float16* sPh = sP + h * 256;
#pragma unroll
      for (int r = 0; r < 8; ++r) {
        float sv = s[r];
        float rmax = sv;
        rmax = fmaxf(rmax, __shfl_xor(rmax, 1));
        rmax = fmaxf(rmax, __shfl_xor(rmax, 2));
        rmax = fmaxf(rmax, __shfl_xor(rmax, 4));
        rmax = fmaxf(rmax, __shfl_xor(rmax, 8));
        float mnew = fmaxf(m[h][r], rmax);
        float corr = __expf(m[h][r] - mnew);
        float p    = __expf(sv - mnew);
        float rs = p;
        rs += __shfl_xor(rs, 1);
        rs += __shfl_xor(rs, 2);
        rs += __shfl_xor(rs, 4);
        rs += __shfl_xor(rs, 8);
        l[h][r]   = l[h][r] * corr + rs;
        acc[h][r] = acc[h][r] * corr;
        m[h][r]   = mnew;
        sPh[(rbase + r) * 16 + col] = (_Float16)p;
      }
    }
    __syncthreads();
#pragma unroll
    for (int h = 0; h < 4; ++h)
      acc[h] = WMMA_F16(frag_a_pad16(sP + h * 256, 16, 0, lane),
                        frag_b_km_pad16(sV, 64, h * 16, lane), acc[h]);
  }
  __syncthreads();
#pragma unroll
  for (int h = 0; h < 4; ++h)
#pragma unroll
    for (int r = 0; r < 8; ++r)
      sX[(rbase + r) * 64 + h * 16 + col] = (_Float16)(acc[h][r] / l[h][r]);
  __syncthreads();

  // ---- out projection + residual (into sO)
  {
    v16h a0 = frag_a(sX, 64, 0, lane);
    v16h a1 = frag_a(sX, 64, 32, lane);
#pragma unroll
    for (int t = 0; t < 4; ++t) {
      v8f c = {};
      c = WMMA_F16(a0, frag_b_wt(owh, 64, t * 16, 0, lane), c);
      c = WMMA_F16(a1, frag_b_wt(owh, 64, t * 16, 32, lane), c);
      float obv = ob[t * 16 + col];
#pragma unroll
      for (int r = 0; r < 8; ++r)
        sO[(rbase + r) * 64 + t * 16 + col] =
            c[r] + obv + sXf[(rbase + r) * 64 + t * 16 + col];
    }
  }
  __syncthreads();

  // ---- layernorm(n12): x1' -> sXf (f32) and sX (f16)
  if (lane < 16) {
    const int row = lane;
    float mean = 0.0f;
    for (int c = 0; c < 64; ++c) mean += sO[row * 64 + c];
    mean *= (1.0f / 64.0f);
    float var = 0.0f;
    for (int c = 0; c < 64; ++c) { float d = sO[row * 64 + c] - mean; var += d * d; }
    var *= (1.0f / 64.0f);
    const float rstd = rsqrtf(var + 1e-5f);
    for (int c = 0; c < 64; ++c) {
      float xn = (sO[row * 64 + c] - mean) * rstd * g12[c] + b12[c];
      sXf[row * 64 + c] = xn;
      sX[row * 64 + c]  = (_Float16)xn;
    }
  }
  __syncthreads();

  // ---- FF1: hidden = gelu(x1' @ W1^T + b1)  [16 x 1024]
  {
    v16h a0 = frag_a(sX, 64, 0, lane);
    v16h a1 = frag_a(sX, 64, 32, lane);
    for (int t = 0; t < 64; ++t) {
      v8f c = {};
      c = WMMA_F16(a0, frag_b_wt(l1h, 64, t * 16, 0, lane), c);
      c = WMMA_F16(a1, frag_b_wt(l1h, 64, t * 16, 32, lane), c);
      float bv = l1b[t * 16 + col];
#pragma unroll
      for (int r = 0; r < 8; ++r) {
        float xv = c[r] + bv;
        sH[(rbase + r) * 1024 + t * 16 + col] = (_Float16)gelu_exact(xv);
      }
    }
  }
  __syncthreads();

  // ---- FF2 + final residual -> global output
  {
    float* dst = out + ((size_t)bb * Lq + q0) * 64;
#pragma unroll
    for (int t = 0; t < 4; ++t) {
      v8f c = {};
      for (int kc = 0; kc < 1024; kc += 32)
        c = WMMA_F16(frag_a(sH, 1024, kc, lane),
                     frag_b_wt(l2h, 1024, t * 16, kc, lane), c);
      float bv = l2b[t * 16 + col];
#pragma unroll
      for (int r = 0; r < 8; ++r)
        dst[(rbase + r) * 64 + t * 16 + col] =
            c[r] + bv + sXf[(rbase + r) * 64 + t * 16 + col];
    }
  }
}

// ---------------------------------------------------------------------------
__global__ void f32_to_f16_kernel(const float* __restrict__ src,
                                  _Float16* __restrict__ dst, int n) {
  int i = blockIdx.x * blockDim.x + threadIdx.x;
  if (i < n) dst[i] = (_Float16)src[i];
}

__global__ void __launch_bounds__(64) max_kernel(const float* __restrict__ f,
                                                 float* __restrict__ out, int L) {
  const int b = blockIdx.x, c = threadIdx.x;
  float m = -1e30f;
  for (int i = 0; i < L; ++i) m = fmaxf(m, f[((size_t)b * L + i) * 64 + c]);
  out[b * 64 + c] = m;
}

// ---------------------------------------------------------------------------
extern "C" void kernel_launch(void* const* d_in, const int* in_sizes, int n_in,
                              void* d_out, int out_size, void* d_ws, size_t ws_size,
                              hipStream_t stream) {
  const int B = 32, N0 = 2048, L1 = 512, L2 = 256;

  const float* xyz = nullptr;
  const float *mW1[3], *mb1[3], *mW2[3], *mb2[3];
  const float *projW[2], *projb[2], *inW[2], *inb[2], *outW[2], *outb[2];
  const float *l1W[2], *l1b[2], *l2W[2], *l2b[2];
  const float *n12g[2], *n12b[2], *n13g[2], *n13b[2];

  if (in_sizes[0] == B * N0 * 3) {
    // insertion order: xyz; mlp0..2 {W1,b1,W2,b2}; atten1..2 {proj_W,proj_b,in_W,in_b,out_W,out_b,l1_W,l1_b,l2_W,l2_b,n12_g,n12_b,n13_g,n13_b}
    int i = 0;
    xyz = (const float*)d_in[i++];
    for (int m = 0; m < 3; ++m) {
      mW1[m] = (const float*)d_in[i++]; mb1[m] = (const float*)d_in[i++];
      mW2[m] = (const float*)d_in[i++]; mb2[m] = (const float*)d_in[i++];
    }
    for (int a = 0; a < 2; ++a) {
      projW[a] = (const float*)d_in[i++]; projb[a] = (const float*)d_in[i++];
      inW[a]   = (const float*)d_in[i++]; inb[a]   = (const float*)d_in[i++];
      outW[a]  = (const float*)d_in[i++]; outb[a]  = (const float*)d_in[i++];
      l1W[a]   = (const float*)d_in[i++]; l1b[a]   = (const float*)d_in[i++];
      l2W[a]   = (const float*)d_in[i++]; l2b[a]   = (const float*)d_in[i++];
      n12g[a]  = (const float*)d_in[i++]; n12b[a]  = (const float*)d_in[i++];
      n13g[a]  = (const float*)d_in[i++]; n13b[a]  = (const float*)d_in[i++];
    }
  } else {
    // jax tree-sorted order: atten1, atten2 (keys sorted: in_W,in_b,l1_W,l1_b,l2_W,l2_b,
    // n12_b,n12_g,n13_b,n13_g,out_W,out_b,proj_W,proj_b), mlp0..2 {W1,W2,b1,b2}, xyz
    int i = 0;
    for (int a = 0; a < 2; ++a) {
      inW[a]   = (const float*)d_in[i++]; inb[a]   = (const float*)d_in[i++];
      l1W[a]   = (const float*)d_in[i++]; l1b[a]   = (const float*)d_in[i++];
      l2W[a]   = (const float*)d_in[i++]; l2b[a]   = (const float*)d_in[i++];
      n12b[a]  = (const float*)d_in[i++]; n12g[a]  = (const float*)d_in[i++];
      n13b[a]  = (const float*)d_in[i++]; n13g[a]  = (const float*)d_in[i++];
      outW[a]  = (const float*)d_in[i++]; outb[a]  = (const float*)d_in[i++];
      projW[a] = (const float*)d_in[i++]; projb[a] = (const float*)d_in[i++];
    }
    for (int m = 0; m < 3; ++m) {
      mW1[m] = (const float*)d_in[i++]; mW2[m] = (const float*)d_in[i++];
      mb1[m] = (const float*)d_in[i++]; mb2[m] = (const float*)d_in[i++];
    }
    xyz = (const float*)d_in[i++];
  }

  // ---- workspace carve
  char* base = (char*)d_ws;
  size_t off = 0;
  auto alloc_f = [&](size_t n) -> float* {
    float* p = (float*)(base + off);
    off += ((n * 4) + 255) & ~(size_t)255;
    return p;
  };
  auto alloc_h = [&](size_t n) -> _Float16* {
    _Float16* p = (_Float16*)(base + off);
    off += ((n * 2) + 255) & ~(size_t)255;
    return p;
  };

  float* lvl1 = alloc_f((size_t)B * L1 * 3);
  float* lvl2 = alloc_f((size_t)B * L2 * 3);
  float* f0   = alloc_f((size_t)B * N0 * 64);
  float* f1   = alloc_f((size_t)B * L1 * 64);
  float* f2   = alloc_f((size_t)B * L2 * 64);
  float* y0   = alloc_f((size_t)B * N0 * 64);
  float* y1q  = alloc_f((size_t)B * L1 * 64);
  float* f1n  = alloc_f((size_t)B * L1 * 64);
  float* y1k  = alloc_f((size_t)B * L1 * 64);
  float* y2q  = alloc_f((size_t)B * L2 * 64);
  float* f2n  = alloc_f((size_t)B * L2 * 64);
  _Float16* k0 = alloc_h((size_t)B * N0 * 64);
  _Float16* v0 = alloc_h((size_t)B * N0 * 64);
  _Float16* k1 = alloc_h((size_t)B * L1 * 64);
  _Float16* v1 = alloc_h((size_t)B * L1 * 64);
  _Float16 *wqh[2], *owh[2], *l1h[2], *l2h[2];
  for (int a = 0; a < 2; ++a) {
    wqh[a] = alloc_h(64 * 64);
    owh[a] = alloc_h(64 * 64);
    l1h[a] = alloc_h(1024 * 64);
    l2h[a] = alloc_h(64 * 1024);
  }

  // ---- pipeline
  fps_kernel<<<B, 256, 0, stream>>>(xyz, lvl1, lvl2);

  mlp_kernel<<<(B * N0 + 255) / 256, 256, 0, stream>>>(xyz,  mW1[0], mb1[0], mW2[0], mb2[0], f0, B * N0);
  mlp_kernel<<<(B * L1 + 255) / 256, 256, 0, stream>>>(lvl1, mW1[1], mb1[1], mW2[1], mb2[1], f1, B * L1);
  mlp_kernel<<<(B * L2 + 255) / 256, 256, 0, stream>>>(lvl2, mW1[2], mb1[2], mW2[2], mb2[2], f2, B * L2);

  for (int a = 0; a < 2; ++a) {
    f32_to_f16_kernel<<<(4096  + 255) / 256, 256, 0, stream>>>(inW[a],  wqh[a], 4096);   // Wq = rows 0..63
    f32_to_f16_kernel<<<(4096  + 255) / 256, 256, 0, stream>>>(outW[a], owh[a], 4096);
    f32_to_f16_kernel<<<(65536 + 255) / 256, 256, 0, stream>>>(l1W[a],  l1h[a], 65536);
    f32_to_f16_kernel<<<(65536 + 255) / 256, 256, 0, stream>>>(l2W[a],  l2h[a], 65536);
  }

  // attention block 1: queries f1 (512), keys f0 (2048)
  proj_ln_kernel<<<(B * N0 + 255) / 256, 256, 0, stream>>>(f0, projW[0], projb[0], n13g[0], n13b[0], y0,  B * N0);
  proj_ln_kernel<<<(B * L1 + 255) / 256, 256, 0, stream>>>(f1, projW[0], projb[0], n13g[0], n13b[0], y1q, B * L1);
  kv_kernel<<<(B * N0 + 255) / 256, 256, 0, stream>>>(y0, inW[0], inb[0], k0, v0, B * N0);
  attn_kernel<<<B * (L1 / 16), 32, 0, stream>>>(y1q, k0, v0, wqh[0], inb[0], owh[0], outb[0],
                                                n12g[0], n12b[0], l1h[0], l1b[0], l2h[0], l2b[0],
                                                f1n, L1, N0);

  // attention block 2: queries f2 (256), keys f1n (512)
  proj_ln_kernel<<<(B * L1 + 255) / 256, 256, 0, stream>>>(f1n, projW[1], projb[1], n13g[1], n13b[1], y1k, B * L1);
  proj_ln_kernel<<<(B * L2 + 255) / 256, 256, 0, stream>>>(f2,  projW[1], projb[1], n13g[1], n13b[1], y2q, B * L2);
  kv_kernel<<<(B * L1 + 255) / 256, 256, 0, stream>>>(y1k, inW[1], inb[1], k1, v1, B * L1);
  attn_kernel<<<B * (L2 / 16), 32, 0, stream>>>(y2q, k1, v1, wqh[1], inb[1], owh[1], outb[1],
                                                n12g[1], n12b[1], l1h[1], l1b[1], l2h[1], l2b[1],
                                                f2n, L2, L1);

  // final max-pool over queries
  max_kernel<<<B, 64, 0, stream>>>(f2n, (float*)d_out, L2);
  (void)n_in; (void)out_size; (void)ws_size;
}
